// MultiProtoAsConv_34419867910619
// MI455X (gfx1250) — matrix operand order repaired and checked by reading, not compiled, not run on hardware
//
#include <hip/hip_runtime.h>
#include <hip/hip_bf16.h>

typedef __attribute__((ext_vector_type(16))) _Float16 v16h;
typedef __attribute__((ext_vector_type(8)))  _Float16 v8h;
typedef __attribute__((ext_vector_type(8)))  float    v8f;

#define C_DIM 256
#define H_DIM 96
#define W_DIM 96
#define HW    9216
#define KH 6
#define KW 6
#define P_LOC 256          // local prototypes (16x16 grid)
#define P_PAD 272          // 257 protos padded to 17*16 rows
#define B_DIM 8
#define TILE_PIX 64        // pixels per workgroup in the main kernel
#define QROW 272           // LDS row stride (halves): 544 B = 17*32 -> 32B-aligned rows

// ---------------------------------------------------------------------------
// K1: pooled mask / bg -> per-prototype softmax bias (0 valid, -1e30 invalid),
//     plus sum(sup_mask) for the global prototype denominator.
// ---------------------------------------------------------------------------
__global__ void prep_masks_kernel(const float* __restrict__ supMask,
                                  const float* __restrict__ bgMask,
                                  float* __restrict__ biasArr,
                                  float* __restrict__ maskSum) {
  __shared__ float red[256];
  const int b = blockIdx.x;
  const int p = threadIdx.x;            // cell index 0..255
  const int gy = p >> 4, gx = p & 15;
  const float* mb = supMask + (size_t)b * HW;
  const float* gb = bgMask  + (size_t)b * HW;
  float ms = 0.f, bs = 0.f;
  for (int i = 0; i < KH; ++i) {
    int base = (gy * KH + i) * W_DIM + gx * KW;
    for (int j = 0; j < KW; ++j) { ms += mb[base + j]; bs += gb[base + j]; }
  }
  const float lm = ms * (1.0f / 36.0f);
  const float lb = bs * (1.0f / 36.0f);
  const bool valid = (lm > 0.95f) && (lb < 0.5f);
  biasArr[b * P_PAD + p] = valid ? 0.0f : -1e30f;

  float part = 0.f;
  for (int k = p; k < HW; k += 256) part += mb[k];
  red[p] = part;
  __syncthreads();
  for (int s = 128; s > 0; s >>= 1) {
    if (p < s) red[p] += red[p + s];
    __syncthreads();
  }
  if (p == 0) { maskSum[b] = red[0]; biasArr[b * P_PAD + 256] = 0.0f; }  // global proto always valid
  if (p >= 1 && p < 16) biasArr[b * P_PAD + 256 + p] = -1e30f;           // padding rows invalid
}

// ---------------------------------------------------------------------------
// K2: one block per (row, batch): build prototype row (avg-pool cell or global
//     masked average), L2-normalize over C, store f16 row-major.
// ---------------------------------------------------------------------------
__global__ void prep_protos_kernel(const float* __restrict__ supFts,
                                   const float* __restrict__ supMask,
                                   const float* __restrict__ maskSum,
                                   _Float16* __restrict__ protoF16) {
  __shared__ float red[256];
  const int row = blockIdx.x;           // 0..271
  const int b   = blockIdx.y;
  const int c   = threadIdx.x;          // channel
  const float* fb = supFts + ((size_t)b * C_DIM + c) * HW;

  float val = 0.0f;
  if (row < P_LOC) {
    const int gy = row >> 4, gx = row & 15;
    float s = 0.f;
    for (int i = 0; i < KH; ++i) {
      int base = (gy * KH + i) * W_DIM + gx * KW;
      for (int j = 0; j < KW; ++j) s += fb[base + j];
    }
    val = s * (1.0f / 36.0f);
  } else if (row == P_LOC) {            // global masked-average prototype
    const float* mb = supMask + (size_t)b * HW;
    float s = 0.f;
    for (int k = 0; k < HW; ++k) s += fb[k] * mb[k];
    val = s / (maskSum[b] + 1e-5f);
  }                                      // rows 257..271: zeros

  red[c] = val * val;
  __syncthreads();
  for (int s = 128; s > 0; s >>= 1) {
    if (c < s) red[c] += red[c + s];
    __syncthreads();
  }
  const float rn = 1.0f / fmaxf(sqrtf(red[0]), 1e-4f);   // safe_norm
  protoF16[((size_t)b * P_PAD + row) * C_DIM + c] = (_Float16)(val * rn);
}

// ---------------------------------------------------------------------------
// K3: fused normalize + WMMA GEMM + online masked softmax + weighted sum.
//     One block = 64 pixels (4 waves x 16-pixel columns), loops 17 proto tiles.
// ---------------------------------------------------------------------------
union AF { v16h v; v8h g[2]; };

__global__ void __launch_bounds__(128)
proto_gemm_softmax_kernel(const float* __restrict__ qry,
                          const _Float16* __restrict__ protoF16,
                          const float* __restrict__ biasArr,
                          float* __restrict__ out) {
  __shared__ __align__(32) _Float16 qtile[TILE_PIX * QROW];
  __shared__ float red[128];
  __shared__ float rnorm[TILE_PIX];
  __shared__ float biasSh[P_PAD];

  const int t = threadIdx.x;
  const int b = blockIdx.y;
  const int pixBase = blockIdx.x * TILE_PIX;

  // -------- Phase 1: stage 64 pixels x 256 channels, per-pixel 1/||q|| -----
  {
    const int p = t & 63;
    const int h = t >> 6;                               // channel half 0/1
    const float* qp = qry + ((size_t)b * C_DIM + h * 128) * HW + pixBase + p;
    float ss = 0.f;
    for (int k = 0; k < 128; ++k) {
      float v = qp[(size_t)k * HW];                      // coalesced across lanes
      ss += v * v;
      qtile[p * QROW + h * 128 + k] = (_Float16)v;
    }
    red[t] = ss;
  }
  __syncthreads();
  if (t < TILE_PIX) {
    float n2 = red[t] + red[t + 64];
    rnorm[t] = 1.0f / fmaxf(sqrtf(n2), 1e-4f);          // safe_norm
  }
  for (int i = t; i < P_PAD; i += 128) biasSh[i] = biasArr[b * P_PAD + i];
  __syncthreads();
  {
    const int p = t & 63;
    const int h = t >> 6;
    const float r = rnorm[p];
    for (int j = 0; j < 16; ++j) {                      // normalize in place
      v8h* ptr = (v8h*)&qtile[p * QROW + h * 128 + j * 8];
      v8h v = *ptr;
      #pragma unroll
      for (int e = 0; e < 8; ++e) v[e] = (_Float16)((float)v[e] * r);
      *ptr = v;
    }
  }
  __syncthreads();

  // -------- Phase 2: WMMA over 17 proto tiles + online masked softmax ------
  const int w    = t >> 5;        // wave -> 16-pixel column
  const int lane = t & 31;
  const int n    = lane & 15;     // A row (M) and B column (N)
  const int hi   = lane >> 4;     // K-group select
  const int pTile = w * 16 + n;

  float m = -3.0e38f, s = 0.f, ta = 0.f;

  for (int mt = 0; mt < 17; ++mt) {
    const _Float16* aRow = protoF16 + ((size_t)b * P_PAD + mt * 16 + n) * C_DIM;
    v8f acc = {};
    #pragma unroll
    for (int k0 = 0; k0 < C_DIM; k0 += 32) {
      AF a;
      // 16-bit A layout: lanes0-15 K={0..7,16..23}, lanes16-31 K={8..15,24..31}
      a.g[0] = *(const v8h*)(aRow + k0 + hi * 8);
      a.g[1] = *(const v8h*)(aRow + k0 + 16 + hi * 8);
      // B layout: same as A, lanes0-15 K={0..7,16..23}, lanes16-31 K={8..15,24..31}
      AF bq;
      bq.g[0] = *(const v8h*)&qtile[pTile * QROW + k0 + hi * 8];
      bq.g[1] = *(const v8h*)&qtile[pTile * QROW + k0 + 16 + hi * 8];
      v16h bb = bq.v;
      acc = __builtin_amdgcn_wmma_f32_16x16x32_f16(
          false, a.v, false, bb, (short)0, acc, false, false);
    }
    // lane owns rows M = mt*16 + hi*8 + r of this pixel's dist column
    float l[8];
    float tileMax = -3.0e38f;
    #pragma unroll
    for (int r = 0; r < 8; ++r) {
      l[r] = acc[r] + biasSh[mt * 16 + hi * 8 + r];
      tileMax = fmaxf(tileMax, l[r]);
    }
    const float mNew  = fmaxf(m, tileMax);
    const float scale = __expf(m - mNew);
    s *= scale; ta *= scale;
    #pragma unroll
    for (int r = 0; r < 8; ++r) {
      float e = __expf(l[r] - mNew);
      s  += e;
      ta += e * acc[r];                 // weighted by raw dist (weight 0 if invalid)
    }
    m = mNew;
  }

  // merge the two lanes holding this pixel (hi=0: M%16<8, hi=1: M%16>=8)
  const float mo = __shfl_xor(m, 16, 32);
  const float so = __shfl_xor(s, 16, 32);
  const float to = __shfl_xor(ta, 16, 32);
  const float M  = fmaxf(m, mo);
  const float S  = s  * __expf(m - M) + so * __expf(mo - M);
  const float T  = ta * __expf(m - M) + to * __expf(mo - M);
  if (hi == 0) out[(size_t)b * HW + pixBase + pTile] = T / S;
}

// ---------------------------------------------------------------------------
extern "C" void kernel_launch(void* const* d_in, const int* in_sizes, int n_in,
                              void* d_out, int out_size, void* d_ws, size_t ws_size,
                              hipStream_t stream) {
  const float* qry  = (const float*)d_in[0];   // qry_fts  [8,256,96,96]
  const float* sup  = (const float*)d_in[1];   // sup_fts  [8,256,96,96]
  const float* mask = (const float*)d_in[2];   // sup_mask [8,1,96,96]
  const float* bg   = (const float*)d_in[3];   // true_bg  [8,1,96,96]
  float* out = (float*)d_out;                  // [8,1,96,96]

  char* ws = (char*)d_ws;
  _Float16* protoF16 = (_Float16*)ws;                               // 8*272*256*2 = 1,114,112 B
  float* biasArr = (float*)(ws + (size_t)B_DIM * P_PAD * C_DIM * 2); // 8*272*4
  float* maskSum = biasArr + B_DIM * P_PAD;                          // 8*4

  prep_masks_kernel<<<B_DIM, 256, 0, stream>>>(mask, bg, biasArr, maskSum);
  prep_protos_kernel<<<dim3(P_PAD, B_DIM), 256, 0, stream>>>(sup, mask, maskSum, protoF16);
  proto_gemm_softmax_kernel<<<dim3(HW / TILE_PIX, B_DIM), 128, 0, stream>>>(
      qry, protoF16, biasArr, out);
  (void)in_sizes; (void)n_in; (void)out_size; (void)ws_size;
}
